// MixtralSparseMoeBlockLora_58360015618612
// MI455X (gfx1250) — matrix operand order repaired and checked by reading, not compiled, not run on hardware
//
#include <hip/hip_runtime.h>
#include <hip/hip_bf16.h>

// ---------------------------------------------------------------------------
// Problem constants (B,S,H,F,E,R,K) = (2,1024,2048,7168,8,32,2)
// ---------------------------------------------------------------------------
#define DIM_B 2
#define DIM_S 1024
#define DIM_H 2048
#define DIM_F 7168
#define DIM_E 8
#define DIM_R 32
#define DIM_K 2
#define DIM_T (DIM_B * DIM_S)   // 2048 tokens
#define DIM_ER (DIM_E * DIM_R)  // 256

typedef __attribute__((ext_vector_type(16))) __bf16 v16bf;
typedef __attribute__((ext_vector_type(8)))  float  v8f;
typedef __attribute__((ext_vector_type(8)))  unsigned short u16x8;

union FragU { v16bf v; u16x8 h[2]; };

__device__ __forceinline__ unsigned short f32_to_bf16_rne(float f) {
    union { float f; unsigned u; } v; v.f = f;
    unsigned r = v.u + 0x7FFFu + ((v.u >> 16) & 1u);
    return (unsigned short)(r >> 16);
}

// ---------------------------------------------------------------------------
// CDNA5 async copy: global -> LDS, 16 bytes per lane, tracked by ASYNCcnt.
// VDST VGPR carries the per-lane LDS byte address (hardware adds LDS_BASE;
// flat-address low 32 bits are the LDS offset per the aperture rules).
// ---------------------------------------------------------------------------
__device__ __forceinline__ void async_b128_to_lds(void* lds_dst, const void* src) {
    unsigned lds_off = (unsigned)(unsigned long long)lds_dst;
    asm volatile("global_load_async_to_lds_b128 %0, %1, off"
                 :: "v"(lds_off), "v"(src)
                 : "memory");
}

__device__ __forceinline__ void wait_asynccnt_zero() {
    asm volatile("s_wait_asynccnt 0" ::: "memory");
}

// ---------------------------------------------------------------------------
// fp32 -> bf16 conversion (grid-stride)
// ---------------------------------------------------------------------------
__global__ void cvt_f32_bf16_kernel(const float* __restrict__ in,
                                    unsigned short* __restrict__ out,
                                    long n) {
    long i = (long)blockIdx.x * blockDim.x + threadIdx.x;
    long stride = (long)gridDim.x * blockDim.x;
    for (; i < n; i += stride) out[i] = f32_to_bf16_rne(in[i]);
}

// ---------------------------------------------------------------------------
// zero-fill fp32 buffer (grid-stride)
// ---------------------------------------------------------------------------
__global__ void zero_f32_kernel(float* __restrict__ out, long n) {
    long i = (long)blockIdx.x * blockDim.x + threadIdx.x;
    long stride = (long)gridDim.x * blockDim.x;
    for (; i < n; i += stride) out[i] = 0.0f;
}

// ---------------------------------------------------------------------------
// Gate: logits = x @ Wg^T, softmax, top-2, renormalized weights.
// One thread per token (tiny: T=2048, E=8).
// ---------------------------------------------------------------------------
__global__ void gate_topk_kernel(const float* __restrict__ x,
                                 const float* __restrict__ Wg,
                                 int* __restrict__ sel,
                                 float* __restrict__ rw) {
    int t = blockIdx.x * blockDim.x + threadIdx.x;
    if (t >= DIM_T) return;
    const float* xr = x + (size_t)t * DIM_H;

    float logit[DIM_E];
    for (int e = 0; e < DIM_E; ++e) {
        const float* wr = Wg + (size_t)e * DIM_H;
        float acc = 0.0f;
        for (int h = 0; h < DIM_H; ++h) acc += xr[h] * wr[h];
        logit[e] = acc;
    }
    float m = logit[0];
    for (int e = 1; e < DIM_E; ++e) m = fmaxf(m, logit[e]);
    float p[DIM_E], psum = 0.0f;
    for (int e = 0; e < DIM_E; ++e) { p[e] = __expf(logit[e] - m); psum += p[e]; }
    float inv = 1.0f / psum;
    for (int e = 0; e < DIM_E; ++e) p[e] *= inv;

    // top-2 (first index wins ties, like lax.top_k)
    int i0 = 0;
    for (int e = 1; e < DIM_E; ++e) if (p[e] > p[i0]) i0 = e;
    int i1 = (i0 == 0) ? 1 : 0;
    for (int e = 0; e < DIM_E; ++e)
        if (e != i0 && p[e] > p[i1]) i1 = e;

    float w0 = p[i0], w1 = p[i1];
    float s = 1.0f / (w0 + w1);
    sel[t * DIM_K + 0] = i0;
    sel[t * DIM_K + 1] = i1;
    rw[t * DIM_K + 0] = w0 * s;
    rw[t * DIM_K + 1] = w1 * s;
}

// ---------------------------------------------------------------------------
// WMMA bf16 GEMM (NT):  C[M,N] (fp32) = A[M,Kd] * B[N,Kd]^T
// A, B row-major bf16 (K contiguous).  Kd % 32 == 0, M % 64 == 0, N % 64 == 0.
//
// Block = 128 threads = 4 waves, 64x64 block tile.
//   - B tile (64 rows x 32 K, 4 KB) is staged to LDS with
//     global_load_async_to_lds_b128 (double buffered, ASYNCcnt + barrier),
//     shared by all 4 waves.
//   - Each wave owns a 16x64 slice: 4 accumulators, 4 WMMAs per K-step,
//     A fragment (direct global, per-wave rows) reused 4x.
// Fragment layouts per CDNA5 ISA §7.12.2:
//   A 16x32: lane L row=L&15, K-chunks [b..b+7] and [16+b..16+b+7], b=8*(L>>4)
//   B 32x16: lane L col=L&15, same K-chunks as A: [b..b+7] and [16+b..16+b+7]
//   C 16x16: element v -> row = 8*(L>>4)+v, col = L&15
// If rowscale != null:  C[row,col] += rowscale[row*scaleStride] * acc
// else:                 C[row,col]  = acc
// ---------------------------------------------------------------------------
__global__ void __launch_bounds__(128)
wmma_gemm_nt_kernel(const unsigned short* __restrict__ A,
                    const unsigned short* __restrict__ B,
                    float* __restrict__ C,
                    int M, int N, int Kd,
                    const float* __restrict__ rowscale, int scaleStride) {
    // double-buffered B tile: [buf][row j (64)][K bytes (64)]
    __shared__ __align__(16) unsigned char ldsB[2][64 * 64];

    const int tid  = threadIdx.x;
    const int lane = tid & 31;
    const int wave = tid >> 5;
    const int m0 = blockIdx.y * 64 + wave * 16;
    const int n0 = blockIdx.x * 64;

    const int mrow  = m0 + (lane & 15);
    const int abase = (lane >> 4) * 8;               // A per-lane chunk offset (elems)
    const unsigned short* Arow = A + (size_t)mrow * Kd;

    // staging plan: 256 chunks of 16B per K-step; 2 chunks per thread
    const int j0 = tid >> 2,          q0 = tid & 3;          // chunk tid
    const int j1 = (tid + 128) >> 2,  q1 = (tid + 128) & 3;  // chunk tid+128

    v8f acc[4] = {v8f{}, v8f{}, v8f{}, v8f{}};

    auto stage = [&](int buf, int kk) {
        async_b128_to_lds(&ldsB[buf][j0 * 64 + q0 * 16],
                          B + (size_t)(n0 + j0) * Kd + kk + q0 * 8);
        async_b128_to_lds(&ldsB[buf][j1 * 64 + q1 * 16],
                          B + (size_t)(n0 + j1) * Kd + kk + q1 * 8);
    };

    stage(0, 0);
    int parity = 0;
    for (int kk = 0; kk < Kd; kk += 32) {
        wait_asynccnt_zero();   // my staged writes for this buffer are in LDS
        __syncthreads();        // everyone's writes visible; prior reads retired

        if (kk + 32 < Kd) stage(parity ^ 1, kk + 32);

        FragU a;
        a.h[0] = *(const u16x8*)(Arow + kk + abase);
        a.h[1] = *(const u16x8*)(Arow + kk + 16 + abase);

        const unsigned char* bb = &ldsB[parity][0];
        #pragma unroll
        for (int s = 0; s < 4; ++s) {
            FragU b;
            const unsigned char* bp =
                bb + (s * 16 + (lane & 15)) * 64 + (lane >> 4) * 16;
            b.h[0] = *(const u16x8*)(bp);
            b.h[1] = *(const u16x8*)(bp + 32);
            acc[s] = __builtin_amdgcn_wmma_f32_16x16x32_bf16(
                         false, a.v, false, b.v, (short)0, acc[s], false, false);
        }
        parity ^= 1;
    }

    const int rbase = m0 + (lane >> 4) * 8;
    if (rowscale == nullptr) {
        #pragma unroll
        for (int s = 0; s < 4; ++s) {
            const int col = n0 + s * 16 + (lane & 15);
            #pragma unroll
            for (int v = 0; v < 8; ++v)
                C[(size_t)(rbase + v) * N + col] = acc[s][v];
        }
    } else {
        #pragma unroll
        for (int s = 0; s < 4; ++s) {
            const int col = n0 + s * 16 + (lane & 15);
            #pragma unroll
            for (int v = 0; v < 8; ++v) {
                float sc = rowscale[(size_t)(rbase + v) * scaleStride];
                C[(size_t)(rbase + v) * N + col] += sc * acc[s][v];
            }
        }
    }
}

// ---------------------------------------------------------------------------
// Up epilogue for slot k:
//   e = sel[t,k]
//   x1 = base1[t,f] + B1[e,f,:]·a1all[t,e,:]
//   x3 = base3[t,f] + B3[e,f,:]·a3all[t,e,:]
//   x2 = silu(x1) * x3   -> bf16
// One block per token; LoRA-a vectors staged in LDS.
// ---------------------------------------------------------------------------
__global__ void up_epilogue_kernel(const float* __restrict__ base1,
                                   const float* __restrict__ base3,
                                   const float* __restrict__ a1all,
                                   const float* __restrict__ a3all,
                                   const float* __restrict__ B1,
                                   const float* __restrict__ B3,
                                   const int* __restrict__ sel,
                                   unsigned short* __restrict__ x2out,
                                   int k) {
    const int t = blockIdx.x;
    const int e = sel[t * DIM_K + k];
    __shared__ float a1s[DIM_R];
    __shared__ float a3s[DIM_R];
    if (threadIdx.x < DIM_R) {
        a1s[threadIdx.x] = a1all[(size_t)t * DIM_ER + e * DIM_R + threadIdx.x];
        a3s[threadIdx.x] = a3all[(size_t)t * DIM_ER + e * DIM_R + threadIdx.x];
    }
    __syncthreads();

    for (int f = threadIdx.x; f < DIM_F; f += blockDim.x) {
        const float* b1r = B1 + ((size_t)e * DIM_F + f) * DIM_R;
        const float* b3r = B3 + ((size_t)e * DIM_F + f) * DIM_R;
        float l1 = 0.0f, l3 = 0.0f;
        #pragma unroll
        for (int r = 0; r < DIM_R; ++r) {
            l1 += b1r[r] * a1s[r];
            l3 += b3r[r] * a3s[r];
        }
        float x1 = base1[(size_t)t * DIM_F + f] + l1;
        float x3 = base3[(size_t)t * DIM_F + f] + l3;
        float silu = x1 / (1.0f + __expf(-x1));
        x2out[(size_t)t * DIM_F + f] = f32_to_bf16_rne(silu * x3);
    }
}

// ---------------------------------------------------------------------------
// LoRA-2 epilogue (both slots):
//   out[t,h] += rw[t,0]*B2[e0,h,:]·a2all0[t,e0,:] + rw[t,1]*B2[e1,h,:]·a2all1[t,e1,:]
// ---------------------------------------------------------------------------
__global__ void lora2_epilogue_kernel(const float* __restrict__ a2all0,
                                      const float* __restrict__ a2all1,
                                      const float* __restrict__ B2,
                                      const int* __restrict__ sel,
                                      const float* __restrict__ rw,
                                      float* __restrict__ out) {
    const int t = blockIdx.x;
    const int e0 = sel[t * DIM_K + 0];
    const int e1 = sel[t * DIM_K + 1];
    const float w0 = rw[t * DIM_K + 0];
    const float w1 = rw[t * DIM_K + 1];
    __shared__ float a0s[DIM_R];
    __shared__ float a1s[DIM_R];
    if (threadIdx.x < DIM_R) {
        a0s[threadIdx.x] = a2all0[(size_t)t * DIM_ER + e0 * DIM_R + threadIdx.x];
        a1s[threadIdx.x] = a2all1[(size_t)t * DIM_ER + e1 * DIM_R + threadIdx.x];
    }
    __syncthreads();

    for (int h = threadIdx.x; h < DIM_H; h += blockDim.x) {
        const float* b0 = B2 + ((size_t)e0 * DIM_H + h) * DIM_R;
        const float* b1 = B2 + ((size_t)e1 * DIM_H + h) * DIM_R;
        float l0 = 0.0f, l1 = 0.0f;
        #pragma unroll
        for (int r = 0; r < DIM_R; ++r) {
            l0 += b0[r] * a0s[r];
            l1 += b1[r] * a1s[r];
        }
        out[(size_t)t * DIM_H + h] += w0 * l0 + w1 * l1;
    }
}

// ---------------------------------------------------------------------------
// Host launcher
// ---------------------------------------------------------------------------
extern "C" void kernel_launch(void* const* d_in, const int* in_sizes, int n_in,
                              void* d_out, int out_size, void* d_ws, size_t ws_size,
                              hipStream_t stream) {
    (void)in_sizes; (void)n_in; (void)out_size; (void)ws_size;

    const float* x  = (const float*)d_in[0];   // (T, H)
    const float* Wg = (const float*)d_in[1];   // (E, H)
    const float* W1 = (const float*)d_in[2];   // (F, H)
    const float* W2 = (const float*)d_in[3];   // (H, F)
    const float* W3 = (const float*)d_in[4];   // (F, H)
    const float* A1 = (const float*)d_in[5];   // (E, R, H) == (ER, H)
    const float* B1 = (const float*)d_in[6];   // (E, F, R)
    const float* A2 = (const float*)d_in[7];   // (E, R, F) == (ER, F)
    const float* B2 = (const float*)d_in[8];   // (E, H, R)
    const float* A3 = (const float*)d_in[9];   // (E, R, H)
    const float* B3 = (const float*)d_in[10];  // (E, F, R)
    float* out = (float*)d_out;                // (T, H)

    // --- workspace carve-out (256B aligned) ---
    char* p = (char*)d_ws;
    auto carve = [&](size_t bytes) -> void* {
        void* r = (void*)p;
        p += (bytes + 255) & ~(size_t)255;
        return r;
    };
    unsigned short* xb    = (unsigned short*)carve((size_t)DIM_T * DIM_H * 2);
    unsigned short* w1b   = (unsigned short*)carve((size_t)DIM_F * DIM_H * 2);
    unsigned short* w3b   = (unsigned short*)carve((size_t)DIM_F * DIM_H * 2);
    unsigned short* w2b   = (unsigned short*)carve((size_t)DIM_H * DIM_F * 2);
    unsigned short* a1rb  = (unsigned short*)carve((size_t)DIM_ER * DIM_H * 2);
    unsigned short* a3rb  = (unsigned short*)carve((size_t)DIM_ER * DIM_H * 2);
    unsigned short* a2rb  = (unsigned short*)carve((size_t)DIM_ER * DIM_F * 2);
    float* base1   = (float*)carve((size_t)DIM_T * DIM_F * 4);
    float* base3   = (float*)carve((size_t)DIM_T * DIM_F * 4);
    float* a1all   = (float*)carve((size_t)DIM_T * DIM_ER * 4);
    float* a3all   = (float*)carve((size_t)DIM_T * DIM_ER * 4);
    unsigned short* x2b0 = (unsigned short*)carve((size_t)DIM_T * DIM_F * 2);
    unsigned short* x2b1 = (unsigned short*)carve((size_t)DIM_T * DIM_F * 2);
    float* a2all0  = (float*)carve((size_t)DIM_T * DIM_ER * 4);
    float* a2all1  = (float*)carve((size_t)DIM_T * DIM_ER * 4);
    int*   sel     = (int*)carve((size_t)DIM_T * DIM_K * 4);
    float* rw      = (float*)carve((size_t)DIM_T * DIM_K * 4);

    auto cvt = [&](const float* src, unsigned short* dst, long n) {
        long blocks = (n + 255) / 256;
        if (blocks > 8192) blocks = 8192;
        cvt_f32_bf16_kernel<<<dim3((unsigned)blocks), dim3(256), 0, stream>>>(src, dst, n);
    };

    // 1) precision conversion of GEMM operands
    cvt(x,  xb,   (long)DIM_T * DIM_H);
    cvt(W1, w1b,  (long)DIM_F * DIM_H);
    cvt(W3, w3b,  (long)DIM_F * DIM_H);
    cvt(W2, w2b,  (long)DIM_H * DIM_F);
    cvt(A1, a1rb, (long)DIM_ER * DIM_H);
    cvt(A3, a3rb, (long)DIM_ER * DIM_H);
    cvt(A2, a2rb, (long)DIM_ER * DIM_F);

    // 2) gate + top-2
    gate_topk_kernel<<<dim3(DIM_T / 256), dim3(256), 0, stream>>>(x, Wg, sel, rw);

    // 3) dense up-projections: base1 = x@W1^T, base3 = x@W3^T   (T x F, Kd = H)
    {
        dim3 grid(DIM_F / 64, DIM_T / 64);
        wmma_gemm_nt_kernel<<<grid, dim3(128), 0, stream>>>(
            xb, w1b, base1, DIM_T, DIM_F, DIM_H, nullptr, 0);
        wmma_gemm_nt_kernel<<<grid, dim3(128), 0, stream>>>(
            xb, w3b, base3, DIM_T, DIM_F, DIM_H, nullptr, 0);
    }

    // 4) LoRA-A projections for all experts: (T x ER, Kd = H)
    {
        dim3 grid(DIM_ER / 64, DIM_T / 64);
        wmma_gemm_nt_kernel<<<grid, dim3(128), 0, stream>>>(
            xb, a1rb, a1all, DIM_T, DIM_ER, DIM_H, nullptr, 0);
        wmma_gemm_nt_kernel<<<grid, dim3(128), 0, stream>>>(
            xb, a3rb, a3all, DIM_T, DIM_ER, DIM_H, nullptr, 0);
    }

    // 5) per-slot LoRA-B + SiLU*mul -> x2 (bf16)
    up_epilogue_kernel<<<dim3(DIM_T), dim3(256), 0, stream>>>(
        base1, base3, a1all, a3all, B1, B3, sel, x2b0, 0);
    up_epilogue_kernel<<<dim3(DIM_T), dim3(256), 0, stream>>>(
        base1, base3, a1all, a3all, B1, B3, sel, x2b1, 1);

    // 6) a2_all per slot: x2 @ A2^T  (T x ER, Kd = F)
    {
        dim3 grid(DIM_ER / 64, DIM_T / 64);
        wmma_gemm_nt_kernel<<<grid, dim3(128), 0, stream>>>(
            x2b0, a2rb, a2all0, DIM_T, DIM_ER, DIM_F, nullptr, 0);
        wmma_gemm_nt_kernel<<<grid, dim3(128), 0, stream>>>(
            x2b1, a2rb, a2all1, DIM_T, DIM_ER, DIM_F, nullptr, 0);
    }

    // 7) out = sum_k rw[t,k] * (x2_k @ W2^T)  (T x H, Kd = F), accumulated in-place
    {
        long n = (long)DIM_T * DIM_H;
        zero_f32_kernel<<<dim3(1024), dim3(256), 0, stream>>>(out, n);
        dim3 grid(DIM_H / 64, DIM_T / 64);
        wmma_gemm_nt_kernel<<<grid, dim3(128), 0, stream>>>(
            x2b0, w2b, out, DIM_T, DIM_H, DIM_F, rw + 0, DIM_K);
        wmma_gemm_nt_kernel<<<grid, dim3(128), 0, stream>>>(
            x2b1, w2b, out, DIM_T, DIM_H, DIM_F, rw + 1, DIM_K);
    }

    // 8) LoRA-2 correction, weighted, both slots
    lora2_epilogue_kernel<<<dim3(DIM_T), dim3(256), 0, stream>>>(
        a2all0, a2all1, B2, sel, rw, out);
}
